// DeepSNN_19636590478088
// MI455X (gfx1250) — matrix operand order repaired and checked by reading, not compiled, not run on hardware
//
#include <hip/hip_runtime.h>
#include <hip/hip_bf16.h>

typedef __attribute__((ext_vector_type(16))) _Float16 v16h;
typedef __attribute__((ext_vector_type(8)))  float    v8f;

union AF16 { v16h v; _Float16 h[16]; };

// ---------------- constants ----------------
#define TT   15
#define CIN  6
#define HW0  160
#define HP   164      // padded input for conv1 (pad 2)
#define C1   128
#define C2   256
#define SH   83       // padded pooled map (pool->81, pad 1)
#define OH   82       // conv2 output side
#define NPIX1 (HW0*HW0) // 25600
#define NPIX2 (OH*OH)   // 6724
#define KC1  5        // ceil(150/32)
#define KC2  16       // 512/32

static const size_t P_HALVES   = (size_t)TT*CIN*HP*HP;   // 2,420,640
static const size_t W1S_HALVES = (size_t)KC1*8*32*16;    // 20,480
static const size_t W2S_HALVES = (size_t)KC2*16*32*16;   // 131,072
static const size_t FT1_N      = (size_t)C1*NPIX1;       // 3,276,800
static const size_t FTIN_N     = (size_t)C1*SH*SH;       // 881,792
static const size_t TOT_N      = (size_t)C2*NPIX2;       // 1,721,344

// compile-time im2col byte offsets (relative to per-lane pixel base)
constexpr int a1_off(int k) {               // conv1: k -> (c,ky,kx), f16 input
    return (k >= CIN * 25) ? -1
        : (((k / 25) * HP + ((k % 25) / 5)) * HP + ((k % 25) % 5)) * 2;
}
constexpr int a2_off(int k) {               // conv2: k = c*4+ky*2+kx, u32 fire-times
    return (((k >> 2) * SH + ((k >> 1) & 1)) * SH + (k & 1)) * 4;
}

// ---------------- init: fire times + big ----------------
__global__ void k_init(unsigned* ft1, unsigned* big, int n) {
    int i = blockIdx.x * blockDim.x + threadIdx.x;
    if (i < n) ft1[i] = 16u;
    if (i == 0) *big = 0u;
}

// ---------------- pad input, f32 -> f16 ----------------
__global__ void k_pad_inp(const float* __restrict__ inp, _Float16* __restrict__ P, int n) {
    int i = blockIdx.x * blockDim.x + threadIdx.x;
    if (i >= n) return;
    int xx = i % HP;
    int t2 = i / HP;
    int yy = t2 % HP;
    int tc = t2 / HP;           // [0, 15*6)
    int x = xx - 2, y = yy - 2;
    float v = 0.f;
    if ((unsigned)x < (unsigned)HW0 && (unsigned)y < (unsigned)HW0)
        v = inp[((size_t)tc * HW0 + y) * HW0 + x];
    P[i] = (_Float16)v;
}

// ---------------- swizzle w1 into WMMA B-fragment blocks ----------------
// layout: W1s[((kc*8 + ntile)*32 + lane)*16 + j],  k = kc*32 + 8*hi + j + (j>=8 ? 8 : 0)
__global__ void k_pack_w1(const float* __restrict__ w1, _Float16* __restrict__ W1s, int n) {
    int i = blockIdx.x * blockDim.x + threadIdx.x;
    if (i >= n) return;
    int j    = i & 15;
    int lane = (i >> 4) & 31;
    int nt   = (i >> 9) & 7;
    int kc   = i >> 12;
    int lo = lane & 15, hi = lane >> 4;
    int nch = nt * 16 + lo;
    int k = kc * 32 + hi * 8 + j + ((j >= 8) ? 8 : 0);
    float v = 0.f;
    if (k < CIN * 25) {
        int c = k / 25; int rem = k - c * 25; int ky = rem / 5; int kx = rem - ky * 5;
        v = w1[(((size_t)nch * CIN + c) * 5 + ky) * 5 + kx];
    }
    W1s[i] = (_Float16)v;
}

// ---------------- swizzle w2 ----------------
// layout: W2s[((kc*16 + ntile)*32 + lane)*16 + j],  k = kc*32 + 8*hi + j + (j>=8 ? 8 : 0)
__global__ void k_pack_w2(const float* __restrict__ w2, _Float16* __restrict__ W2s, int n) {
    int i = blockIdx.x * blockDim.x + threadIdx.x;
    if (i >= n) return;
    int j    = i & 15;
    int lane = (i >> 4) & 31;
    int nt   = (i >> 9) & 15;
    int kc   = i >> 13;
    int lo = lane & 15, hi = lane >> 4;
    int nch = nt * 16 + lo;
    int k = kc * 32 + hi * 8 + j + ((j >= 8) ? 8 : 0);  // < 512 always
    int c = k >> 2, ky = (k >> 1) & 1, kx = k & 1;
    W2s[i] = (_Float16)w2[(((size_t)nch * C1 + c) * 2 + ky) * 2 + kx];
}

// ---------------- conv1 WMMA GEMM: LDS-staged A, fire-time atomicMin epilogue -------
// grid(1600, 15), block 256 = 8 waves; wave kc builds A chunk kc, wave wv computes ntile wv
__global__ __launch_bounds__(256) void k_conv1(const _Float16* __restrict__ P,
                                               const _Float16* __restrict__ W1s,
                                               unsigned* __restrict__ ft1) {
    __shared__ v16h As[KC1 * 32];
    int tid  = threadIdx.x;
    int lane = tid & 31, wv = tid >> 5;
    int lo   = lane & 15, hi = lane >> 4;
    int mt   = blockIdx.x;                   // pixel tile 0..1599
    int t    = blockIdx.y;                   // time

    // ---- cooperative A staging (constant-folded im2col offsets) ----
    {
        int m  = mt * 16 + lo;
        int py = m / HW0, px = m - py * HW0;
        const char* base = (const char*)(P + (size_t)t * CIN * HP * HP)
                         + (size_t)(py * HP + px) * 2;
#pragma unroll
        for (int kc = 0; kc < KC1; ++kc) {
            if (wv == kc) {                  // uniform per-wave branch
                AF16 frag;
#pragma unroll
                for (int j = 0; j < 16; ++j) {
                    const int k0 = kc * 32 + j + ((j >= 8) ? 8 : 0);  // hi=0 k
                    const int o0 = a1_off(k0);
                    const int o1 = a1_off(k0 + 8);                    // hi=1 k
                    _Float16 v = (_Float16)0.f;
                    if (o0 >= 0 && o1 >= 0) {
                        v = *(const _Float16*)(base + (hi ? o1 : o0));
                    } else if (o0 >= 0 || o1 >= 0) {
                        int o = hi ? o1 : o0;
                        if (o >= 0) v = *(const _Float16*)(base + o);
                    }
                    frag.h[j] = v;
                }
                As[kc * 32 + lane] = frag.v;
            }
        }
    }
    __syncthreads();

    // ---- compute: 5 chained WMMAs, B fragments are single contiguous v16h loads ----
    v8f acc = {};
#pragma unroll
    for (int kc = 0; kc < KC1; ++kc) {
        v16h a = As[kc * 32 + lane];
        v16h b = *(const v16h*)(W1s + (((size_t)(kc * 8 + wv) * 32 + lane) * 16));
        acc = __builtin_amdgcn_wmma_f32_16x16x32_f16(false, a, false, b,
                                                     (short)0, acc, false, false);
    }

    // ---- epilogue: ft1 is [c][25600] flat, pixel index == mm (no div needed) ----
    int ch = wv * 16 + lo;                   // D layout: n = lane&15
    unsigned* ftch = ft1 + (size_t)ch * NPIX1 + (size_t)(mt * 16 + hi * 8);
#pragma unroll
    for (int r = 0; r < 8; ++r) {
        if (acc[r] >= 10.0f)                 // THR1 spike at time t
            atomicMin(ftch + r, (unsigned)t);
    }
}

// ---------------- maxpool(2,2,pad1) + pad1 + pointwise inhibition on fire times ----
__global__ void k_pool_inhib(const unsigned* __restrict__ ft1,
                             unsigned* __restrict__ ftin, int n) {   // n = 83*83
    int i = blockIdx.x * blockDim.x + threadIdx.x;
    if (i >= n) return;
    int h = i / SH, w = i - (i / SH) * SH;
    unsigned bestft = 16u; int winner = -1;
    if (h >= 1 && h <= 81 && w >= 1 && w <= 81) {
        int r = h - 1, cx = w - 1;
        int y0 = 2 * r - 1, x0 = 2 * cx - 1;
        for (int c = 0; c < C1; ++c) {
            const unsigned* F = ft1 + (size_t)c * NPIX1;
            unsigned f = 16u;
            for (int dy = 0; dy < 2; ++dy) {
                int yy = y0 + dy; if ((unsigned)yy >= (unsigned)HW0) continue;
                for (int dx = 0; dx < 2; ++dx) {
                    int xx = x0 + dx; if ((unsigned)xx >= (unsigned)HW0) continue;
                    unsigned q = F[yy * HW0 + xx];
                    if (q < f) f = q;
                }
            }
            if (f < bestft) { bestft = f; winner = c; }  // strict < => lowest channel on tie
        }
    }
    for (int c = 0; c < C1; ++c) {
        unsigned out = 16u;
        if (c == winner && bestft < 16u) out = bestft;
        ftin[(size_t)c * (SH * SH) + i] = out;
    }
}

// ---------------- conv2 WMMA GEMM: LDS-staged A built from fire-time compares ------
// grid(421, 15), block 512 = 16 waves; wave kc builds chunk kc, wave wv computes ntile wv
__global__ __launch_bounds__(512) void k_conv2(const unsigned* __restrict__ ftin,
                                               const _Float16* __restrict__ W2s,
                                               float* __restrict__ vout) {
    __shared__ v16h As[KC2 * 32];
    int tid  = threadIdx.x;
    int lane = tid & 31, wv = tid >> 5;      // wv = 0..15
    int lo   = lane & 15, hi = lane >> 4;
    int mt   = blockIdx.x;                   // 0..420
    int t    = blockIdx.y;

    // ---- cooperative A staging ----
    {
        int m = mt * 16 + lo;
        if (m > NPIX2 - 1) m = NPIX2 - 1;    // clamp: garbage rows feed unstored D rows
        int py = m / OH, px = m - py * OH;
        const char* base = (const char*)ftin + (size_t)(py * SH + px) * 4;
        unsigned tu = (unsigned)t;
#pragma unroll
        for (int kc = 0; kc < KC2; ++kc) {
            if (wv == kc) {                  // uniform per-wave branch
                AF16 frag;
#pragma unroll
                for (int j = 0; j < 16; ++j) {
                    const int k0 = kc * 32 + j + ((j >= 8) ? 8 : 0);
                    const int o0 = a2_off(k0);
                    const int o1 = a2_off(k0 + 8);
                    unsigned f = *(const unsigned*)(base + (hi ? o1 : o0));
                    frag.h[j] = (f <= tu) ? (_Float16)1.f : (_Float16)0.f;
                }
                As[kc * 32 + lane] = frag.v;
            }
        }
    }
    __syncthreads();

    // ---- compute: 16 chained WMMAs ----
    v8f acc = {};
#pragma unroll
    for (int kc = 0; kc < KC2; ++kc) {
        v16h a = As[kc * 32 + lane];
        v16h b = *(const v16h*)(W2s + (((size_t)(kc * 16 + wv) * 32 + lane) * 16));
        acc = __builtin_amdgcn_wmma_f32_16x16x32_f16(false, a, false, b,
                                                     (short)0, acc, false, false);
    }

    int ch = wv * 16 + lo;
    float* vch = vout + ((size_t)t * C2 + ch) * NPIX2 + (size_t)(mt * 16 + hi * 8);
    int mbase = mt * 16 + hi * 8;
#pragma unroll
    for (int r = 0; r < 8; ++r) {
        if (mbase + r < NPIX2) {
            float v = acc[r];
            v = (v < 1.0f) ? 0.0f : v;       // fire(THR2)
            vch[r] = v;
        }
    }
}

// ---------------- second pointwise inhibition on real-valued v; writes s ----------
__global__ void k_inhib2(float* __restrict__ s_out, float* __restrict__ v_out, int n) {
    int i = blockIdx.x * blockDim.x + threadIdx.x;
    if (i >= n) return;                      // i = pixel in 82*82
    int widx[TT];
    int clampsum = 0, lastclamp = 0;
    for (int t = 0; t < TT; ++t) {
        float mx = v_out[((size_t)t * C2 + 0) * NPIX2 + i]; int am = 0;
        for (int c = 1; c < C2; ++c) {
            float vv = v_out[((size_t)t * C2 + c) * NPIX2 + i];
            if (vv > mx) { mx = vv; am = c; }         // strict > => argmax first occurrence
        }
        widx[t] = am;
        int cl = (mx > 0.f) ? 1 : 0;
        clampsum += cl;
        if (t == TT - 1) lastclamp = cl;
    }
    int earliest = TT - clampsum;
    if (earliest < 0) earliest = 0;
    if (earliest > TT - 1) earliest = TT - 1;
    int winner = widx[earliest];
    for (int t = 0; t < TT; ++t)
        for (int c = 0; c < C2; ++c) {
            size_t idx = ((size_t)t * C2 + c) * NPIX2 + i;
            float vv = v_out[idx];
            float o = (lastclamp && c == winner) ? vv : 0.f;
            v_out[idx] = o;
            s_out[idx] = (o > 0.f) ? 1.f : 0.f;
        }
}

// ---------------- big = max(s*vals) via nonneg-float bit atomicMax ----------------
__global__ void k_bigprep(const float* __restrict__ s_out, const float* __restrict__ v_out,
                          unsigned* __restrict__ big, int n) {   // n = 256*6724
    int i = blockIdx.x * blockDim.x + threadIdx.x;
    if (i >= n) return;
    int nspk = 0;
    for (int t = 0; t < TT; ++t) nspk += (s_out[(size_t)t * n + i] > 0.f) ? 1 : 0;
    if (nspk > 0) {
        int e = TT - nspk; if (e < 0) e = 0; if (e > TT - 1) e = TT - 1;
        float vals = v_out[(size_t)e * n + i];    // >= 0, bit pattern order-preserving
        atomicMax(big, __float_as_uint(vals));
    }
}

// ---------------- total[c,h,w] = nspk * (vals + big*T) ----------------
__global__ void k_total(const float* __restrict__ s_out, const float* __restrict__ v_out,
                        const unsigned* __restrict__ big, float* __restrict__ total, int n) {
    int i = blockIdx.x * blockDim.x + threadIdx.x;
    if (i >= n) return;
    int nspk = 0;
    for (int t = 0; t < TT; ++t) nspk += (s_out[(size_t)t * n + i] > 0.f) ? 1 : 0;
    int e = TT - nspk; if (e < 0) e = 0; if (e > TT - 1) e = TT - 1;
    float vals = v_out[(size_t)e * n + i];
    float bg = __uint_as_float(*big) * (float)TT;
    total[i] = (float)nspk * (vals + bg);
}

// ---------------- greedy k-WTA with spatial NMS (single block) ----------------
__global__ __launch_bounds__(1024) void k_kwta(float* __restrict__ total,
                                               float* __restrict__ wout) {
    __shared__ float sv[1024];
    __shared__ int   si[1024];
    __shared__ float winrec[24];
    const int n = C2 * NPIX2;
    int tid = threadIdx.x;
    for (int it = 0; it < 8; ++it) {
        float bv = -1.f; int bi = 0;
        for (int i = tid; i < n; i += 1024) {
            float v = total[i];
            if (v > bv) { bv = v; bi = i; }          // ascending scan keeps lowest index
        }
        sv[tid] = bv; si[tid] = bi;
        __syncthreads();
        for (int stp = 512; stp > 0; stp >>= 1) {
            if (tid < stp) {
                float ov = sv[tid + stp]; int oi = si[tid + stp];
                if (ov > sv[tid] || (ov == sv[tid] && oi < si[tid])) { sv[tid] = ov; si[tid] = oi; }
            }
            __syncthreads();
        }
        float mv = sv[0]; int idx = si[0];
        int active = (mv != 0.f);
        int f = idx / NPIX2, rem = idx - f * NPIX2;
        int r = rem / OH, col = rem - r * OH;
        if (tid == 0) {
            winrec[it * 3 + 0] = active ? (float)f   : -1.f;
            winrec[it * 3 + 1] = active ? (float)r   : -1.f;
            winrec[it * 3 + 2] = active ? (float)col : -1.f;
        }
        __syncthreads();
        if (active) {
            for (int i = tid; i < NPIX2; i += 1024) total[f * NPIX2 + i] = 0.f;   // kill feature
            for (int i = tid; i < 9 * C2; i += 1024) {                            // kill radius
                int cc = i / 9, p = i - cc * 9;
                int rr = r + (p / 3) - 1, c2 = col + (p % 3) - 1;
                if ((unsigned)rr < (unsigned)OH && (unsigned)c2 < (unsigned)OH)
                    total[cc * NPIX2 + rr * OH + c2] = 0.f;
            }
        }
        __syncthreads();
    }
    if (tid < 24) wout[tid] = winrec[tid];
}

static inline size_t align256(size_t x) { return (x + 255) & ~(size_t)255; }

extern "C" void kernel_launch(void* const* d_in, const int* in_sizes, int n_in,
                              void* d_out, int out_size, void* d_ws, size_t ws_size,
                              hipStream_t stream) {
    const float* inp = (const float*)d_in[0];
    const float* w1  = (const float*)d_in[1];
    const float* w2  = (const float*)d_in[2];
    // d_in[3] = layer_idx (always 2) -> deepest branch, ignored

    float* out = (float*)d_out;
    const size_t SZ = (size_t)TT * C2 * NPIX2;
    float* s_out = out;
    float* v_out = out + SZ;
    float* w_out = out + 2 * SZ;

    char* ws = (char*)d_ws;
    size_t off = 0;
    _Float16* P    = (_Float16*)(ws + off); off = align256(off + P_HALVES * 2);
    _Float16* W1s  = (_Float16*)(ws + off); off = align256(off + W1S_HALVES * 2);
    _Float16* W2s  = (_Float16*)(ws + off); off = align256(off + W2S_HALVES * 2);
    unsigned* ft1  = (unsigned*)(ws + off); off = align256(off + FT1_N * 4);
    unsigned* ftin = (unsigned*)(ws + off); off = align256(off + FTIN_N * 4);
    float*    tot  = (float*)   (ws + off); off = align256(off + TOT_N * 4);
    unsigned* big  = (unsigned*)(ws + off); off = align256(off + 4);
    (void)ws_size; (void)in_sizes; (void)n_in; (void)out_size;

    k_init<<<(int)((FT1_N + 255) / 256), 256, 0, stream>>>(ft1, big, (int)FT1_N);
    k_pad_inp<<<(int)((P_HALVES + 255) / 256), 256, 0, stream>>>(inp, P, (int)P_HALVES);
    k_pack_w1<<<(int)((W1S_HALVES + 255) / 256), 256, 0, stream>>>(w1, W1s, (int)W1S_HALVES);
    k_pack_w2<<<(int)((W2S_HALVES + 255) / 256), 256, 0, stream>>>(w2, W2s, (int)W2S_HALVES);

    k_conv1<<<dim3(1600, TT), 256, 0, stream>>>(P, W1s, ft1);
    k_pool_inhib<<<(SH * SH + 255) / 256, 256, 0, stream>>>(ft1, ftin, SH * SH);
    k_conv2<<<dim3(421, TT), 512, 0, stream>>>(ftin, W2s, v_out);

    k_inhib2<<<(NPIX2 + 255) / 256, 256, 0, stream>>>(s_out, v_out, NPIX2);
    k_bigprep<<<(int)((TOT_N + 255) / 256), 256, 0, stream>>>(s_out, v_out, big, (int)TOT_N);
    k_total<<<(int)((TOT_N + 255) / 256), 256, 0, stream>>>(s_out, v_out, big, tot, (int)TOT_N);
    k_kwta<<<1, 1024, 0, stream>>>(tot, w_out);
}
